// Model_1726576853140
// MI455X (gfx1250) — matrix operand order repaired and checked by reading, not compiled, not run on hardware
//
#include <hip/hip_runtime.h>

#define HH   1024
#define WW   1024
#define CC   16
#define HWSZ (HH*WW)
#define TH   16
#define TW   64
#define NPIX (TH*TW)

typedef __attribute__((ext_vector_type(16))) _Float16     v16h;
typedef __attribute__((ext_vector_type(8)))  float        v8f;
typedef __attribute__((ext_vector_type(4)))  unsigned int v4u;
typedef __attribute__((ext_vector_type(8)))  int          v8i;
typedef __attribute__((ext_vector_type(4)))  int          v4i;

// 16-bit A-matrix 16x32 element->K mapping (CDNA5 ISA §7.12.2):
// VGPR v (=e/2), half h (=e%2): v<4 -> K = 2v+h (+8 for lanes 16..31)
//                               v>=4 -> K = 16+2(v-4)+h (+8 for lanes 16..31)
__device__ __forceinline__ int kmap(int e, int hi) {
    int v = e >> 1, h = e & 1;
    int base = (v < 4) ? (2*v + h) : (16 + 2*(v-4) + h);
    return base + (hi ? 8 : 0);
}

__global__ __launch_bounds__(256, 1)
void nca_fused_kernel(const float* __restrict__ state,
                      const float* __restrict__ convw,
                      const float* __restrict__ w1,
                      const float* __restrict__ b1v,
                      const float* __restrict__ w2,
                      const float* __restrict__ b2v,
                      const float* __restrict__ mask,
                      float* __restrict__ sgout) {
    // LDS: state tile + perception features + per-wave staging
    // sTile is fully contiguous [c][y][x] (pitch 66) so the TDM 3D tile
    // (X=66 fastest, Y=18, Z=16) lands exactly on it.
    __shared__ float    sTile[CC][TH+2][TW+2];
    __shared__ _Float16 perc[NPIX][48];              // 48 features / pixel
    __shared__ _Float16 xstage[8][16][128];          // per-wave hidden acts
    __shared__ float    sgstage[8][16][16];          // per-wave sg transpose
    __shared__ float    cw[48*9];                    // conv weights

    const int tid  = threadIdx.x;
    const int lane = tid & 31;
    const int wv   = tid >> 5;
    const int hi   = (lane >= 16) ? 1 : 0;
    const int mrow = lane & 15;

    const int tilesX = WW / TW;
    const int tileX  = (blockIdx.x % tilesX) * TW;
    const int tileY  = (blockIdx.x / tilesX) * TH;

    // ---- stage conv weights ----
    for (int i = tid; i < 48*9; i += 256) cw[i] = convw[i];

    // ---- prefetch rand-mask tile rows (global_prefetch_b8) ----
    {
        int c = tid & 15, r = tid >> 4;
        __builtin_prefetch(&mask[c*HWSZ + (tileY + r)*WW + tileX], 0, 1);
    }

    // ---- stage state tile with 1px halo into LDS ----
#if __has_builtin(__builtin_amdgcn_tensor_load_to_lds)
    const bool interior = (tileX > 0) && (tileY > 0);
    if (interior) {
        // Tensor Data Mover: DMA [16][18][66] f32 tile straight into LDS.
        // tensor_dim0/1 = remaining extent from tile start, so right/bottom
        // halo past the image auto-reads zero (SAME padding). Left/top halo
        // is in-bounds for interior tiles.
        if (wv == 0) {
            const unsigned gx0 = (unsigned)(tileX - 1);
            const unsigned gy0 = (unsigned)(tileY - 1);
            unsigned long long gaddr =
                (unsigned long long)(uintptr_t)state +
                4ull * ((unsigned long long)gy0 * WW + gx0);
            unsigned ldsa = (unsigned)(uintptr_t)&sTile[0][0][0];
            unsigned td0  = (unsigned)WW - gx0;      // remaining x extent
            unsigned td1  = (unsigned)HH - gy0;      // remaining y extent

            // D# group 0: count=1, lds_addr, global_addr[56:0], type=2
            v4u g0 = { 1u,
                       ldsa,
                       (unsigned)(gaddr & 0xFFFFFFFFull),
                       (unsigned)((gaddr >> 32) & 0x1FFFFFFull) | (2u << 30) };
            // D# group 1: data_size=4B; dims/strides in elements
            v8i g1 = { (int)(2u << 16),                               // data_size
                       (int)((td0 & 0xFFFFu) << 16),                  // dim0 lo16
                       (int)(((td0 >> 16) & 0xFFFFu) |
                             ((td1 & 0xFFFFu) << 16)),                // dim0 hi/dim1 lo
                       (int)(((td1 >> 16) & 0xFFFFu) | (66u << 16)),  // dim1 hi | tile_dim0
                       (int)(18u | (16u << 16)),                      // tile_dim1 | tile_dim2
                       (int)WW,                                       // dim0_stride lo32
                       0,                                             // dim0_stride hi | dim1_stride lo16
                       (int)((unsigned)HWSZ >> 16) };                 // dim1_stride [47:16]
            // D# group 2: tensor_dim2=16 (channels), dim2_stride=H*W
            v4i g2 = { (int)CC, 0, (int)HWSZ, 0 };
            v4i g3 = { 0, 0, 0, 0 };
            v8i g4 = { 0, 0, 0, 0, 0, 0, 0, 0 };     // VADDR4 slot: unused
            __builtin_amdgcn_tensor_load_to_lds(g0, g1, g2, g3, g4, 0);
            __builtin_amdgcn_s_wait_tensorcnt(0);
        }
    } else
#endif
    {
        // generic zero-padded fill (image-border tiles)
        for (int i = tid; i < CC*(TH+2)*(TW+2); i += 256) {
            int c  = i / ((TH+2)*(TW+2));
            int r  = i % ((TH+2)*(TW+2));
            int ly = r / (TW+2), lx = r % (TW+2);
            int gy = tileY + ly - 1, gx = tileX + lx - 1;
            float v = 0.f;
            if (gy >= 0 && gy < HH && gx >= 0 && gx < WW)
                v = state[c*HWSZ + gy*WW + gx];
            sTile[c][ly][lx] = v;
        }
    }
    __syncthreads();

    // ---- perception: depthwise 3x3 conv, 3 filters per channel ----
    for (int P = tid; P < NPIX; P += 256) {
        int ly = P / TW, lx = P % TW;
        #pragma unroll
        for (int c = 0; c < CC; ++c) {
            float n[9];
            #pragma unroll
            for (int dy = 0; dy < 3; ++dy)
                #pragma unroll
                for (int dx = 0; dx < 3; ++dx)
                    n[dy*3+dx] = sTile[c][ly+dy][lx+dx];
            #pragma unroll
            for (int f = 0; f < 3; ++f) {
                int o = 3*c + f;
                float s = 0.f;
                #pragma unroll
                for (int j = 0; j < 9; ++j) s += cw[o*9 + j] * n[j];
                perc[P][o] = (_Float16)s;
            }
        }
    }
    __syncthreads();

    // ---- weight B-fragments (lane = N-local col, element = K-local row) ----
    v16h B1[8][2];
    #pragma unroll
    for (int t = 0; t < 8; ++t)
        #pragma unroll
        for (int k = 0; k < 2; ++k)
            #pragma unroll
            for (int e = 0; e < 16; ++e) {
                int i = 32*k + kmap(e, hi);                // K (input feature)
                B1[t][k][e] = (i < 48) ? (_Float16)w1[(16*t + mrow)*48 + i]
                                       : (_Float16)0.f;
            }
    v16h B2[4];
    #pragma unroll
    for (int k = 0; k < 4; ++k)
        #pragma unroll
        for (int e = 0; e < 16; ++e)
            B2[k][e] = (_Float16)w2[mrow*128 + 32*k + kmap(e, hi)];

    float bias1[8];
    #pragma unroll
    for (int t = 0; t < 8; ++t) bias1[t] = b1v[16*t + mrow];
    const float bias2 = b2v[mrow];

    // ---- 8 groups of 16 pixels per wave ----
    for (int i = 0; i < 8; ++i) {
        int g   = wv*8 + i;
        int gy  = g >> 2;
        int px0 = (g & 3) * 16;
        int P0  = gy*TW + px0;

        // A fragments: perception, K=0..31 and K=32..47 (+zero pad)
        v16h a0, a1;
        #pragma unroll
        for (int e = 0; e < 16; ++e) {
            int k0 = kmap(e, hi);
            a0[e] = perc[P0 + mrow][k0];
            int k1 = 32 + k0;
            a1[e] = (k1 < 48) ? perc[P0 + mrow][k1] : (_Float16)0.f;
        }

        // dense1 (48->128) + ReLU, stage as f16 for dense2 A
        #pragma unroll
        for (int t = 0; t < 8; ++t) {
            v8f acc;
            #pragma unroll
            for (int r = 0; r < 8; ++r) acc[r] = bias1[t];
            acc = __builtin_amdgcn_wmma_f32_16x16x32_f16(
                      false, a0, false, B1[t][0], (short)0, acc, false, false);
            acc = __builtin_amdgcn_wmma_f32_16x16x32_f16(
                      false, a1, false, B1[t][1], (short)0, acc, false, false);
            #pragma unroll
            for (int r = 0; r < 8; ++r) {
                float xv = acc[r] > 0.f ? acc[r] : 0.f;
                xstage[wv][r + hi*8][16*t + mrow] = (_Float16)xv;
            }
        }

        // dense2 (128->16)
        v8f acc2;
        #pragma unroll
        for (int r = 0; r < 8; ++r) acc2[r] = bias2;
        #pragma unroll
        for (int k = 0; k < 4; ++k) {
            v16h a2;
            #pragma unroll
            for (int e = 0; e < 16; ++e)
                a2[e] = xstage[wv][mrow][32*k + kmap(e, hi)];
            acc2 = __builtin_amdgcn_wmma_f32_16x16x32_f16(
                       false, a2, false, B2[k], (short)0, acc2, false, false);
        }

        // residual + stochastic mask; transpose through LDS
        int Y = tileY + gy;
        #pragma unroll
        for (int r = 0; r < 8; ++r) {
            int M  = r + hi*8;           // pixel index in group
            int xl = px0 + M;
            int c  = mrow;               // channel (C/D col)
            float st = sTile[c][gy+1][xl+1];
            float mk = mask[c*HWSZ + Y*WW + tileX + xl];
            sgstage[wv][M][c] = st + acc2[r]*mk;
        }
        // coalesced sg store: 2 channels x 16 consecutive x per iteration
        #pragma unroll
        for (int cc = 0; cc < 8; ++cc) {
            int c    = cc*2 + hi;
            int xoff = mrow;
            sgout[c*HWSZ + Y*WW + tileX + px0 + xoff] = sgstage[wv][xoff][c];
        }
    }
}

// pass 2: 3x3 max-pool alive gate on channel 3 (sg resident in 192MB L2)
__global__ __launch_bounds__(256)
void nca_alive_kernel(const float* __restrict__ sg, float* __restrict__ out) {
    int idx = blockIdx.x * 256 + threadIdx.x;
    int x = idx % WW, y = idx / WW;
    const float* alpha = sg + 3*HWSZ;
    float m = -3.4e38f;
    #pragma unroll
    for (int dy = -1; dy <= 1; ++dy) {
        int yy = y + dy;
        if (yy < 0 || yy >= HH) continue;
        #pragma unroll
        for (int dx = -1; dx <= 1; ++dx) {
            int xx = x + dx;
            if (xx < 0 || xx >= WW) continue;
            float v = alpha[yy*WW + xx];
            m = v > m ? v : m;
        }
    }
    float alive = (m > 0.1f) ? 1.f : 0.f;
    #pragma unroll
    for (int c = 0; c < CC; ++c)
        out[c*HWSZ + y*WW + x] = sg[c*HWSZ + y*WW + x] * alive;
}

extern "C" void kernel_launch(void* const* d_in, const int* in_sizes, int n_in,
                              void* d_out, int out_size, void* d_ws, size_t ws_size,
                              hipStream_t stream) {
    const float* state = (const float*)d_in[0];
    const float* convw = (const float*)d_in[1];
    const float* w1    = (const float*)d_in[2];
    const float* b1    = (const float*)d_in[3];
    const float* w2    = (const float*)d_in[4];
    const float* b2    = (const float*)d_in[5];
    const float* mask  = (const float*)d_in[6];
    float* sg  = (float*)d_ws;   // [16,1024,1024] f32 = 64 MB scratch
    float* out = (float*)d_out;

    dim3 blk(256);
    dim3 grid1((HH/TH)*(WW/TW));           // 1024 blocks, 16x64 pixel tiles
    nca_fused_kernel<<<grid1, blk, 0, stream>>>(state, convw, w1, b1, w2, b2,
                                                mask, sg);
    nca_alive_kernel<<<(HH*WW)/256, blk, 0, stream>>>(sg, out);
}
